// VectorQuantizer_25220047962780
// MI455X (gfx1250) — matrix operand order repaired and checked by reading, not compiled, not run on hardware
//
#include <hip/hip_runtime.h>
#include <hip/hip_bf16.h>

typedef __attribute__((ext_vector_type(16))) __bf16 v16bf;
typedef __attribute__((ext_vector_type(8)))  float  v8f;
typedef __attribute__((ext_vector_type(4)))  unsigned int v4u;
typedef __attribute__((ext_vector_type(8)))  int    v8i;
typedef __attribute__((ext_vector_type(4)))  int    v4i;

#define DDIM   256
#define KCODES 256
#define NROWS  262144
#define NPART  (NROWS / 128 * 8)   // one partial per wave: 2048 blocks * 8 waves

// ---------------------------------------------------------------------------
// Setup: convert codebook fp32 -> bf16 (for WMMA B operand) and compute ||e||^2
// ---------------------------------------------------------------------------
__global__ __launch_bounds__(256) void vq_setup(const float* __restrict__ emb,
                                                __hip_bfloat16* __restrict__ emb_bf,
                                                float* __restrict__ e2) {
    const int k = blockIdx.x;
    const int t = threadIdx.x;
    float v = emb[k * DDIM + t];
    ((__bf16*)emb_bf)[k * DDIM + t] = (__bf16)v;

    __shared__ float red[256];
    red[t] = v * v;
    __syncthreads();
    for (int s = 128; s > 0; s >>= 1) {
        if (t < s) red[t] += red[t + s];
        __syncthreads();
    }
    if (t == 0) e2[k] = red[0];
}

// ---------------------------------------------------------------------------
// Main: per block, 128 rows of z_e vs all 256 codes.
//   - bf16 codebook (128KB) DMA'd into LDS by the Tensor Data Mover
//   - each of 8 waves: 16-row A tile, 16 N-tiles x 8 WMMA(16x16x32 bf16)
//   - running argmin of (||e||^2 - 2*dot), butterfly-reduced across 16 lanes
//   - gather fp32 emb row (L2-hot) -> z_q_st output + squared-diff loss partial
// ---------------------------------------------------------------------------
__global__ __launch_bounds__(256) void vq_main(const float* __restrict__ z_e,
                                               const float* __restrict__ emb,
                                               const __hip_bfloat16* __restrict__ emb_bf,
                                               const float* __restrict__ e2,
                                               float* __restrict__ out_zq,
                                               float* __restrict__ out_ind,
                                               float* __restrict__ partials) {
    extern __shared__ char smem[];
    __bf16* s_emb = (__bf16*)smem;                         // 256*256 bf16 = 128KB

    const int tid = threadIdx.x;

    // --- TDM: DMA the whole bf16 codebook (2D tile 256x256 x 2B) into LDS ---
    // D# group0: count=1, lds_addr=0, global_addr, type=2 ("image")
    // D# group1: data_size=2B, tensor_dim0=256, tensor_dim1=256,
    //            tile_dim0=256, tile_dim1=256, tensor_dim0_stride=256
    if (tid < 32) {  // wave 0 issues the DMA (EXEC ignored by TDM)
        unsigned long long ga = (unsigned long long)emb_bf;
        v4u g0;
        g0.x = 1u;                                         // count=1
        g0.y = 0u;                                         // lds_addr = 0
        g0.z = (unsigned int)(ga & 0xFFFFFFFFu);           // global_addr[31:0]
        g0.w = (unsigned int)((ga >> 32) & 0x1FFFFFFu)     // global_addr[56:32]
             | (2u << 30);                                 // type=2
        v8i g1;
        g1[0] = 0x00010000;   // wg_mask=0, data_size=1 (2 bytes)
        g1[1] = 0x01000000;   // tensor_dim0[15:0]=256 in bits [31:16]
        g1[2] = 0x01000000;   // tensor_dim1[15:0]=256 in bits [31:16]
        g1[3] = 0x01000000;   // tile_dim0=256 in bits [31:16]
        g1[4] = 0x00000100;   // tile_dim1=256, tile_dim2=0
        g1[5] = 0x00000100;   // tensor_dim0_stride=256 (low 32)
        g1[6] = 0x00000000;   // stride0 hi / stride1 lo
        g1[7] = 0x00000001;   // tensor_dim1_stride=65536 (bits 47:16)
        v4i g2 = {0, 0, 0, 0};
        v4i g3 = {0, 0, 0, 0};
        v8i g4 = {0, 0, 0, 0, 0, 0, 0, 0};
        __builtin_amdgcn_tensor_load_to_lds(g0, g1, g2, g3, g4, 0);
        __builtin_amdgcn_s_wait_tensorcnt(0);
    }

    const int  wave = tid >> 5;
    const int  lane = tid & 31;
    const int  l16  = lane & 15;
    const bool hi   = lane >= 16;
    const int  row_base = blockIdx.x * 128 + wave * 16;

    // Preload per-lane ||e||^2 residue classes (coalesced, L2-hot) while DMA runs
    float ecache[16];
#pragma unroll
    for (int nt = 0; nt < 16; ++nt) ecache[nt] = e2[nt * 16 + l16];

    // --- Load A tile: 16 rows x 256 K as 8 chunks of v16bf ------------------
    // ISA A layout (16-bit 16x32): lane m = lane%16; lanes 0-15 hold K [0..7]
    // in v0-3 and [16..23] in v4-7; lanes 16-31 hold K [8..15] and [24..31].
    v16bf a[8];
    const float* arow = z_e + (size_t)(row_base + l16) * DDIM;
#pragma unroll
    for (int kc = 0; kc < 8; ++kc) {
        const int klo = kc * 32 + (hi ? 8 : 0);
        float4 f0 = *(const float4*)(arow + klo);
        float4 f1 = *(const float4*)(arow + klo + 4);
        float4 f2 = *(const float4*)(arow + klo + 16);
        float4 f3 = *(const float4*)(arow + klo + 20);
        float tf[16] = {f0.x, f0.y, f0.z, f0.w, f1.x, f1.y, f1.z, f1.w,
                        f2.x, f2.y, f2.z, f2.w, f3.x, f3.y, f3.z, f3.w};
#pragma unroll
        for (int i = 0; i < 16; ++i) a[kc][i] = (__bf16)tf[i];
    }

    __syncthreads();   // codebook now resident in LDS

    // --- Distance sweep over all codes -------------------------------------
    float best_d[8];
    int   best_i[8];
#pragma unroll
    for (int j = 0; j < 8; ++j) { best_d[j] = 3.4e38f; best_i[j] = 0; }

    for (int nt = 0; nt < 16; ++nt) {
        const int code = nt * 16 + l16;
        // B layout (16-bit 32x16): lane holds col = lane%16; lanes 0-15 hold
        // K [0..7],[16..23]; lanes 16-31 hold K [8..15],[24..31] (matches A).
        const __bf16* bbase = s_emb + code * DDIM + (hi ? 8 : 0);
        v8f acc = {};
#pragma unroll
        for (int kc = 0; kc < 8; ++kc) {
            typedef __attribute__((ext_vector_type(8))) __bf16 v8bf_t;
            v8bf_t b0 = *(const v8bf_t*)(bbase + kc * 32);
            v8bf_t b1 = *(const v8bf_t*)(bbase + kc * 32 + 16);
            v16bf b = __builtin_shufflevector(b0, b1, 0, 1, 2, 3, 4, 5, 6, 7, 8, 9, 10, 11, 12, 13, 14, 15);
            acc = __builtin_amdgcn_wmma_f32_16x16x32_bf16(
                false, a[kc], false, b, (short)0, acc, false, false);
        }
        const float ec = ecache[nt];
#pragma unroll
        for (int j = 0; j < 8; ++j) {
            float d = ec - 2.0f * acc[j];      // ||e||^2 - 2 z.e  (argmin-equiv)
            if (d < best_d[j]) { best_d[j] = d; best_i[j] = code; }
        }
    }

    // Butterfly argmin across the 16 lanes holding each row (tie -> lower idx)
#pragma unroll
    for (int j = 0; j < 8; ++j) {
#pragma unroll
        for (int m = 1; m < 16; m <<= 1) {
            float od = __shfl_xor(best_d[j], m, 16);
            int   oi = __shfl_xor(best_i[j], m, 16);
            if (od < best_d[j] || (od == best_d[j] && oi < best_i[j])) {
                best_d[j] = od; best_i[j] = oi;
            }
        }
    }

    // --- Emit indices, z_q_st, and loss partial ----------------------------
    float lsum = 0.0f;
#pragma unroll
    for (int j = 0; j < 8; ++j) {
        const int row  = row_base + j + (hi ? 8 : 0);  // lanes<16: row j; else j+8
        const int code = best_i[j];                     // uniform per 16-lane half
        if (l16 == 0) out_ind[row] = (float)code;

        const float* erow = emb + (size_t)code * DDIM;  // fp32, L2-resident
        const float* zrow = z_e + (size_t)row * DDIM;
        float*       orow = out_zq + (size_t)row * DDIM;
        const int c0 = l16 * 16;                        // 16 lanes x 16 floats
#pragma unroll
        for (int q = 0; q < 4; ++q) {
            float4 e4 = *(const float4*)(erow + c0 + q * 4);
            float4 z4 = *(const float4*)(zrow + c0 + q * 4);
            *(float4*)(orow + c0 + q * 4) = e4;         // z_q_st == z_q forward
            float dx = z4.x - e4.x, dy = z4.y - e4.y;
            float dz = z4.z - e4.z, dw = z4.w - e4.w;
            lsum += dx * dx + dy * dy + dz * dz + dw * dw;
        }
    }

    // Wave-level sum, one partial per wave (deterministic: no float atomics)
#pragma unroll
    for (int m = 1; m < 32; m <<= 1) lsum += __shfl_xor(lsum, m, 32);
    if (lane == 0) partials[blockIdx.x * 8 + wave] = lsum;
}

// ---------------------------------------------------------------------------
// Final deterministic reduction of loss partials:
// loss = (1 + BETA) * sum / (N*D), BETA = 0.25
// ---------------------------------------------------------------------------
__global__ __launch_bounds__(256) void vq_reduce(const float* __restrict__ partials,
                                                 float* __restrict__ out_loss) {
    __shared__ float red[256];
    const int t = threadIdx.x;
    float s = 0.0f;
    for (int i = t; i < NPART; i += 256) s += partials[i];
    red[t] = s;
    __syncthreads();
    for (int k = 128; k > 0; k >>= 1) {
        if (t < k) red[t] += red[t + k];
        __syncthreads();
    }
    if (t == 0)
        out_loss[0] = 1.25f * red[0] / (float)((size_t)NROWS * DDIM);
}

// ---------------------------------------------------------------------------
extern "C" void kernel_launch(void* const* d_in, const int* in_sizes, int n_in,
                              void* d_out, int out_size, void* d_ws, size_t ws_size,
                              hipStream_t stream) {
    const float* z_e = (const float*)d_in[0];   // [N, D] fp32
    const float* emb = (const float*)d_in[1];   // [K, D] fp32

    char* ws = (char*)d_ws;
    __hip_bfloat16* emb_bf = (__hip_bfloat16*)ws;                    // 128KB
    float* e2       = (float*)(ws + (size_t)KCODES * DDIM * 2);      // 1KB
    float* partials = (float*)(ws + (size_t)KCODES * DDIM * 2 + 1024);

    float* out      = (float*)d_out;
    float* out_zq   = out;                                   // [N*D]
    float* out_ind  = out + (size_t)NROWS * DDIM;            // [N] (as float)
    float* out_loss = out_ind + NROWS;                       // [1]

    vq_setup<<<KCODES, 256, 0, stream>>>(emb, emb_bf, e2);

    const size_t shmem = (size_t)KCODES * DDIM * 2;          // 128KB (<320KB/WGP)
    vq_main<<<NROWS / 128, 256, shmem, stream>>>(z_e, emb, emb_bf, e2,
                                                 out_zq, out_ind, partials);

    vq_reduce<<<1, 256, 0, stream>>>(partials, out_loss);
}
